// SNN_Detection_Bottleneck_85444079387268
// MI455X (gfx1250) — matrix operand order repaired and checked by reading, not compiled, not run on hardware
//
#include <hip/hip_runtime.h>

typedef __attribute__((ext_vector_type(16))) _Float16     v16h;
typedef __attribute__((ext_vector_type(8)))  _Float16     v8h;
typedef __attribute__((ext_vector_type(8)))  float        v8f;
typedef __attribute__((ext_vector_type(8)))  unsigned int v8u;

__device__ __forceinline__ float fast_sigmoid(float x) {
    return 1.0f / (1.0f + __expf(-x));
}

union h2u { _Float16 h[2]; unsigned int u; };

// ---------------------------------------------------------------------------
// fp32 [B,C,H,W] -> f16 channel-pair-interleaved padded [B,C/2,H+2,W+2,2]
// with zero halo (one pass)
// ---------------------------------------------------------------------------
__global__ __launch_bounds__(256)
void pad_cvt_kernel(const float* __restrict__ src, _Float16* __restrict__ dst,
                    int Chalf, int H, int W, int n)   // n = B*C*(H+2)*(W+2)
{
    int i = blockIdx.x * blockDim.x + threadIdx.x;
    if (i >= n) return;
    const int Wp = W + 2, Hp = H + 2, HWp = Hp * Wp;
    const int c01 = i & 1;
    const int r   = i >> 1;
    const int pix = r % HWp;
    const int bcp = r / HWp;
    const int cp  = bcp % Chalf;
    const int b   = bcp / Chalf;
    const int c   = cp * 2 + c01;
    const int yy  = pix / Wp;
    const int xx  = pix % Wp;
    _Float16 v = (_Float16)0;
    if (yy >= 1 && yy <= H && xx >= 1 && xx <= W)
        v = (_Float16)src[((b * (2 * Chalf) + c) * H + (yy - 1)) * W + (xx - 1)];
    dst[i] = v;
}

// zero fill (16B per thread); n8 = elements/8
__global__ __launch_bounds__(256)
void zero_f16_kernel(_Float16* __restrict__ dst, int n8)
{
    int i = blockIdx.x * blockDim.x + threadIdx.x;
    if (i >= n8) return;
    ((v8h*)dst)[i] = v8h{};
}

// beta = sigmoid(beta_raw), 36 channels
__global__ void beta_prep_kernel(const float* __restrict__ braw, float* __restrict__ beta)
{
    int i = threadIdx.x;
    if (i < 36) beta[i] = fast_sigmoid(braw[i]);
}

// ---------------------------------------------------------------------------
// Weight swizzle: fp32 [Cout, Cin*9] -> f16 A-fragments [KS][MT][32 lanes]
// A-fragment layout used here (16x32 f16; contiguous k, not the ISA table order):
//   k = e + 16*half
// GEMM K order is tap-major: k = pos*Cin + ci  ->  weight col = ci*9 + pos
// ---------------------------------------------------------------------------
__global__ __launch_bounds__(256)
void swizzle_w_kernel(const float* __restrict__ w, v16h* __restrict__ out,
                      int Cout, int Cin9, int Cin, int MT, int nthreads)
{
    int idx = blockIdx.x * blockDim.x + threadIdx.x;
    if (idx >= nthreads) return;
    int lane = idx & 31;
    int frag = idx >> 5;
    int half = lane >> 4;
    int mm   = lane & 15;
    int mt   = frag % MT;
    int ks   = frag / MT;
    int m    = mt * 16 + mm;
    v16h f;
#pragma unroll
    for (int e = 0; e < 16; ++e) {
        int k = ks * 32 + e + 16 * half;
        float v = 0.0f;
        if (m < Cout && k < Cin9) {
            int pos = k / Cin;
            int ci  = k - pos * Cin;
            v = w[m * Cin9 + ci * 9 + pos];
        }
        f[e] = (_Float16)v;
    }
    out[idx] = f;
}

// ---------------------------------------------------------------------------
// B-fragment layout (32x16 f16, KxN): lanes 0-15: col n=lane, e -> K=e;
// lanes 16-31: col n=lane-16, e -> K=16+e. Register pair j holds K = 2j,2j+1
// -> consecutive channels -> one aligned b32 load from interleaved buffer.
// C/D f32 16x16: col n = lane&15, VGPR r -> row m = r + 8*(lane>>4).
// ---------------------------------------------------------------------------

// Encoder: conv3x3(Cin->36) + BN + LIF(T=4) + mean -> interleaved f16 avg.
template <int Cin, int H, int W>
__global__ __launch_bounds__(256)
void enc_lif_kernel(const unsigned int* __restrict__ featp, // [B,Cin/2,Hp,Wp] pairs
                    const v16h*         __restrict__ wswz,  // [9*Cin/32][3][32]
                    const float*        __restrict__ bn_s,
                    const float*        __restrict__ bn_b,
                    const float*        __restrict__ thr,
                    const float*        __restrict__ beta_t,
                    unsigned int*       __restrict__ avgp)  // [B,18,Hp,Wp] pairs
{
    constexpr int Hp = H + 2, Wp = W + 2, HWp = Hp * Wp;
    constexpr int CS = Cin / 32;
    const int lane = threadIdx.x & 31;
    const int wave = threadIdx.x >> 5;
    const int half = lane >> 4;
    const int ln   = lane & 15;
    const int p    = (blockIdx.x * 8 + wave) * 16 + ln;   // NPIX % 128 == 0
    const int x    = p % W;
    const int y    = (p / W) % H;
    const int b    = p / (H * W);

    // 9 always-valid tap base offsets in pair units; half*8 pairs folded in
    int off[9];
#pragma unroll
    for (int pos = 0; pos < 9; ++pos) {
        const int kh = pos / 3, kw = pos - 3 * (pos / 3);
        off[pos] = (b * (Cin / 2) + half * 8) * HWp + (y + kh) * Wp + (x + kw);
    }

    v8f acc[3];
#pragma unroll
    for (int mt = 0; mt < 3; ++mt) acc[mt] = v8f{};

#pragma unroll
    for (int pos = 0; pos < 9; ++pos) {
        const int o = off[pos];
#pragma unroll
        for (int cs = 0; cs < CS; ++cs) {
            v8u bq;
#pragma unroll
            for (int j = 0; j < 8; ++j)
                bq[j] = featp[o + (cs * 16 + j) * HWp];   // imm offsets
            const v16h bf = __builtin_bit_cast(v16h, bq);
            const int ks = pos * CS + cs;
            const v16h* wp = wswz + (size_t)(ks * 3) * 32 + lane;
#pragma unroll
            for (int mt = 0; mt < 3; ++mt) {
                v16h af = wp[mt * 32];
                acc[mt] = __builtin_amdgcn_wmma_f32_16x16x32_f16(
                    false, af, false, bf, (short)0, acc[mt], false, false);
            }
        }
    }

    // epilogue: affine + LIF (forward spike = hard threshold) + mean,
    // channel pairs packed into one b32 store
    const int pix = (y + 1) * Wp + (x + 1);
#pragma unroll
    for (int mt = 0; mt < 3; ++mt) {
#pragma unroll
        for (int j = 0; j < 4; ++j) {
            const int ch = mt * 16 + 8 * half + 2 * j;    // even
            if (ch < 36) {
                h2u pk;
#pragma unroll
                for (int q = 0; q < 2; ++q) {
                    const int c = ch + q;
                    const float h    = acc[mt][2 * j + q] * bn_s[c] + bn_b[c];
                    const float t    = thr[c];
                    const float beta = beta_t[c];
                    float m = 0.0f, sum = 0.0f;
#pragma unroll
                    for (int ts = 0; ts < 4; ++ts) {
                        m = beta * m + h;
                        if (m > t) { sum += 1.0f; m -= t; }
                    }
                    pk.h[q] = (_Float16)(sum * 0.25f);
                }
                avgp[(b * 18 + (ch >> 1)) * HWp + pix] = pk.u;
            }
        }
    }
}

// Decoder: conv3x3(36->Cout=MT*16) + BN + SiLU -> fp32 output.
// K order pos-major: k = pos*36 + ci, K=324 padded to 352; ks fully unrolled.
// Pairs never cross a tap boundary (36 even) -> all b32 loads.
template <int MT, int H, int W>
__global__ __launch_bounds__(256)
void dec_silu_kernel(const unsigned int* __restrict__ avgp, // [B,18,Hp,Wp] pairs
                     const v16h*         __restrict__ wswz, // [11][MT][32]
                     const float*        __restrict__ bn_s,
                     const float*        __restrict__ bn_b,
                     float*              __restrict__ out)  // [B,Cout,H,W]
{
    constexpr int Hp = H + 2, Wp = W + 2, HWp = Hp * Wp;
    constexpr int Cout = MT * 16;
    constexpr int K = 324;
    const int lane = threadIdx.x & 31;
    const int wave = threadIdx.x >> 5;
    const int half = lane >> 4;
    const int ln   = lane & 15;
    const int p    = (blockIdx.x * 8 + wave) * 16 + ln;
    const int x    = p % W;
    const int y    = (p / W) % H;
    const int b    = p / (H * W);

    int off[9];
#pragma unroll
    for (int pos = 0; pos < 9; ++pos) {
        const int kh = pos / 3, kw = pos - 3 * (pos / 3);
        off[pos] = b * 18 * HWp + (y + kh) * Wp + (x + kw);
    }

    v8f acc[MT];
#pragma unroll
    for (int mt = 0; mt < MT; ++mt) acc[mt] = v8f{};

#pragma unroll
    for (int ks = 0; ks < 11; ++ks) {
        v8u bq;
        if (half == 0) {
#pragma unroll
            for (int j = 0; j < 8; ++j) {
                const int k = ks * 32 + 2 * j;            // compile-time, even
                if (k < K) {
                    const int pos = k / 36, ci = k - 36 * (k / 36);
                    bq[j] = avgp[off[pos] + (ci >> 1) * HWp];
                } else bq[j] = 0u;
            }
        } else {
#pragma unroll
            for (int j = 0; j < 8; ++j) {
                const int k = ks * 32 + 16 + 2 * j;       // compile-time, even
                if (k < K) {
                    const int pos = k / 36, ci = k - 36 * (k / 36);
                    bq[j] = avgp[off[pos] + (ci >> 1) * HWp];
                } else bq[j] = 0u;
            }
        }
        const v16h bf = __builtin_bit_cast(v16h, bq);
        const v16h* wp = wswz + (size_t)(ks * MT) * 32 + lane;
#pragma unroll
        for (int mt = 0; mt < MT; ++mt) {
            v16h af = wp[mt * 32];
            acc[mt] = __builtin_amdgcn_wmma_f32_16x16x32_f16(
                false, af, false, bf, (short)0, acc[mt], false, false);
        }
    }

#pragma unroll
    for (int mt = 0; mt < MT; ++mt) {
#pragma unroll
        for (int r = 0; r < 8; ++r) {
            const int ch = mt * 16 + r + 8 * half;
            const float yv = acc[mt][r] * bn_s[ch] + bn_b[ch];
            out[((b * Cout + ch) * H + y) * W + x] = yv * fast_sigmoid(yv);
        }
    }
}

// ---------------------------------------------------------------------------
// Launch
// ---------------------------------------------------------------------------
extern "C" void kernel_launch(void* const* d_in, const int* in_sizes, int n_in,
                              void* d_out, int out_size, void* d_ws, size_t ws_size,
                              hipStream_t stream)
{
    (void)in_sizes; (void)n_in; (void)out_size; (void)ws_size;

    // workspace layout (all offsets multiples of 256B)
    char* ws = (char*)d_ws;
    unsigned int* a3p = (unsigned int*)(ws + 0);         // 16*18*162*162*4 = 30,233,088
    unsigned int* a4p = (unsigned int*)(ws + 30233088);  // 16*18*82*82*4   =  7,746,048
    unsigned int* a5p = (unsigned int*)(ws + 37979136);  // 16*18*42*42*4   =  2,032,128
    unsigned int* f3p = (unsigned int*)(ws + 40011264);  // 16*32*162*162*4 = 53,747,712
    unsigned int* f4p = (unsigned int*)(ws + 93758976);  // 16*64*82*82*4   = 27,541,504
    unsigned int* f5p = (unsigned int*)(ws + 121300480); // 16*128*42*42*4  = 14,450,688
    v16h* encw3 = (v16h*)(ws + 135751168);               //  55,296
    v16h* encw4 = (v16h*)(ws + 135806464);               // 110,592
    v16h* encw5 = (v16h*)(ws + 135917056);               // 221,184
    v16h* decw3 = (v16h*)(ws + 136138240);               //  45,056
    v16h* decw4 = (v16h*)(ws + 136183296);               //  90,112
    v16h* decw5 = (v16h*)(ws + 136273408);               // 180,224
    float* beta3 = (float*)(ws + 136453632);             // 256
    float* beta4 = (float*)(ws + 136453888);             // 256
    float* beta5 = (float*)(ws + 136454144);             // 256

    const float* feat[3]; const float* encW[3]; const float* encS[3]; const float* encB[3];
    const float* thr[3];  const float* braw[3];
    const float* decW[3]; const float* decS[3]; const float* decB[3];
    for (int l = 0; l < 3; ++l) {
        const int o = l * 9;
        feat[l] = (const float*)d_in[o + 0];
        encW[l] = (const float*)d_in[o + 1];
        encS[l] = (const float*)d_in[o + 2];
        encB[l] = (const float*)d_in[o + 3];
        thr[l]  = (const float*)d_in[o + 4];
        braw[l] = (const float*)d_in[o + 5];
        decW[l] = (const float*)d_in[o + 6];
        decS[l] = (const float*)d_in[o + 7];
        decB[l] = (const float*)d_in[o + 8];
    }

    // 1) pad+convert feats to interleaved f16 halo mirrors; zero padded avg
    {
        int n = 16 * 64 * 162 * 162;
        pad_cvt_kernel<<<(n + 255) / 256, 256, 0, stream>>>(
            feat[0], (_Float16*)f3p, 32, 160, 160, n);
        n = 16 * 128 * 82 * 82;
        pad_cvt_kernel<<<(n + 255) / 256, 256, 0, stream>>>(
            feat[1], (_Float16*)f4p, 64, 80, 80, n);
        n = 16 * 256 * 42 * 42;
        pad_cvt_kernel<<<(n + 255) / 256, 256, 0, stream>>>(
            feat[2], (_Float16*)f5p, 128, 40, 40, n);

        int n8 = 16 * 36 * 162 * 162 / 8;
        zero_f16_kernel<<<(n8 + 255) / 256, 256, 0, stream>>>((_Float16*)a3p, n8);
        n8 = 16 * 36 * 82 * 82 / 8;
        zero_f16_kernel<<<(n8 + 255) / 256, 256, 0, stream>>>((_Float16*)a4p, n8);
        n8 = 16 * 36 * 42 * 42 / 8;
        zero_f16_kernel<<<(n8 + 255) / 256, 256, 0, stream>>>((_Float16*)a5p, n8);
    }

    // 2) weight swizzles (both tap-major) + beta tables
    const int Cin[3] = {64, 128, 256};
    const int KSe[3] = {18, 36, 72};
    const int MTd[3] = {4, 8, 16};
    v16h* encw[3] = {encw3, encw4, encw5};
    v16h* decw[3] = {decw3, decw4, decw5};
    float* beta[3] = {beta3, beta4, beta5};
    for (int l = 0; l < 3; ++l) {
        int ne = KSe[l] * 3 * 32;
        swizzle_w_kernel<<<(ne + 255) / 256, 256, 0, stream>>>(
            encW[l], encw[l], 36, Cin[l] * 9, Cin[l], 3, ne);
        int nd = 11 * MTd[l] * 32;
        swizzle_w_kernel<<<(nd + 255) / 256, 256, 0, stream>>>(
            decW[l], decw[l], MTd[l] * 16, 324, 36, MTd[l], nd);
        beta_prep_kernel<<<1, 64, 0, stream>>>(braw[l], beta[l]);
    }

    // 3) fused enc-conv + BN + LIF + mean (writes interior of padded avg)
    enc_lif_kernel<64, 160, 160><<<3200, 256, 0, stream>>>(
        f3p, encw3, encS[0], encB[0], thr[0], beta3, a3p);
    enc_lif_kernel<128, 80, 80><<<800, 256, 0, stream>>>(
        f4p, encw4, encS[1], encB[1], thr[1], beta4, a4p);
    enc_lif_kernel<256, 40, 40><<<200, 256, 0, stream>>>(
        f5p, encw5, encS[2], encB[2], thr[2], beta5, a5p);

    // 4) fused dec-conv + BN + SiLU -> fp32 (concatenated r3|r4|r5)
    float* out = (float*)d_out;
    dec_silu_kernel<4, 160, 160><<<3200, 256, 0, stream>>>(
        a3p, decw3, decS[0], decB[0], out);
    dec_silu_kernel<8, 80, 80><<<800, 256, 0, stream>>>(
        a4p, decw4, decS[1], decB[1], out + 26214400);
    dec_silu_kernel<16, 40, 40><<<200, 256, 0, stream>>>(
        a5p, decw5, decS[2], decB[2], out + 39321600);
}
